// VectorQuantizerEMA_36017595744529
// MI455X (gfx1250) — matrix operand order repaired and checked by reading, not compiled, not run on hardware
//
#include <hip/hip_runtime.h>
#include <hip/hip_bf16.h>
#include <math.h>

// ---------------- problem constants ----------------
#define B_   32
#define C_   256       // embedding dim D
#define H_   32
#define W_   32
#define HW_  (H_*W_)   // 1024
#define N_   (B_*HW_)  // 32768 flattened vectors
#define K_   1024      // codebook size
#define D_   256
#define NT_  (N_/16)   // 2048 row tiles
#define CT_  (K_/16)   // 64 col tiles
#define DC_  (D_/32)   // 8 K-chunks of 32 (bf16 WMMA K=32)
#define QELEMS_ ((size_t)B_*C_*HW_)   // 8388608

typedef __bf16 v16bf __attribute__((ext_vector_type(16)));
typedef float  v8f   __attribute__((ext_vector_type(8)));

// float -> bf16 round-to-nearest-even (bit pattern)
__device__ __forceinline__ unsigned short f2bf(float x) {
    unsigned u = __float_as_uint(x);
    u += 0x7FFFu + ((u >> 16) & 1u);
    return (unsigned short)(u >> 16);
}

__device__ __forceinline__ float blockReduceSum256(float v) {
    #pragma unroll
    for (int off = 16; off; off >>= 1) v += __shfl_xor(v, off, 32);
    __shared__ float s[8];
    int w = threadIdx.x >> 5, l = threadIdx.x & 31;
    if (l == 0) s[w] = v;
    __syncthreads();
    float r = 0.f;
    if (threadIdx.x == 0) {
        #pragma unroll
        for (int i = 0; i < 8; ++i) r += s[i];
    }
    return r; // valid in thread 0
}

// ---------------- K0: zero accumulators ----------------
__global__ __launch_bounds__(1024)
void k_init(float* __restrict__ colsum, float* __restrict__ kldAcc,
            float* __restrict__ elatAcc) {
    colsum[threadIdx.x] = 0.f;
    if (threadIdx.x == 0) { *kldAcc = 0.f; *elatAcc = 0.f; }
}

// ---------------- K1: embedding -> bf16 B-fragment layout + norms ----------
// B frag (bf16 32x16 KxN): lane = ncol + 16*((dd>>3)&1); e = (dd&7) + 8*(dd>>4)
__global__ __launch_bounds__(256)
void k_prep_emb(const float* __restrict__ emb, unsigned short* __restrict__ embB,
                float* __restrict__ embNorm) {
    const int k = blockIdx.x, d = threadIdx.x;
    const float e = emb[k * D_ + d];
    const int ct = k >> 4, lc = k & 15;
    const int dchunk = d >> 5, dd = d & 31;
    const int lane = lc + (((dd >> 3) & 1) << 4);
    const int eidx = (dd & 7) + ((dd >> 4) << 3);
    embB[(((size_t)(ct * DC_ + dchunk) * 32 + lane) << 4) + eidx] = f2bf(e);
    float s = blockReduceSum256(e * e);
    if (threadIdx.x == 0) embNorm[k] = s;
}

// ---------------- K2: inputs (NCHW) -> bf16 A-fragment layout + row norms --
// A frag (bf16 16x32 MxK): lane = m + 16*((dd>>3)&1); e = (dd&7) + 8*(dd>>4)
__global__ __launch_bounds__(256)
void k_prep_x(const float* __restrict__ x, unsigned short* __restrict__ flatA,
              float* __restrict__ rowNorm) {
    const int n  = blockIdx.x * 256 + threadIdx.x;   // grid = N_/256
    const int b  = n / HW_;
    const int hw = n % HW_;
    const int rt = n >> 4, m = n & 15;
    float s = 0.f;
    for (int d = 0; d < D_; ++d) {
        const float v = x[((size_t)(b * C_ + d)) * HW_ + hw]; // coalesced across lanes
        s += v * v;
        const int dchunk = d >> 5, dd = d & 31;
        const int lane = m + (((dd >> 3) & 1) << 4);
        const int e = (dd & 7) + ((dd >> 4) << 3);
        flatA[(((size_t)(rt * DC_ + dchunk) * 32 + lane) << 4) + e] = f2bf(v);
    }
    rowNorm[n] = s;
}

// ---------------- K3: fused WMMA distances + gumbel-softmax + kld ----------
// 1 block = 16 rows x 1024 codes; 8 waves, wave w owns col tiles 8w..8w+7.
__global__ __launch_bounds__(256)
void k_dist_softmax(const uint4* __restrict__ flatA, const uint4* __restrict__ embB,
                    const float* __restrict__ rowNorm, const float* __restrict__ embNorm,
                    const float* __restrict__ gumbel,
                    float* __restrict__ enc, float* __restrict__ colsum,
                    float* __restrict__ kldAcc,
                    int* __restrict__ idxArr, float* __restrict__ vsel) {
    const int rt = blockIdx.x;
    const int tid = threadIdx.x;
    const int wave = tid >> 5, lane = tid & 31;
    const int half = lane >> 4, lcol = lane & 15;
    const int rowBase = rt * 16;

    __shared__ float sRowN[16];
    __shared__ float sMaxW[8 * 16];
    __shared__ int   sAColW[8 * 16];
    __shared__ float sSumW[8 * 16];
    __shared__ float sMax[16], sSum[16];
    __shared__ int   sIdx[16];
    __shared__ float sKld[8];

    if (tid < 16) sRowN[tid] = rowNorm[rowBase + tid];

    // --- preload A fragments for all 8 K-chunks (registers, 64 VGPRs) ---
    v16bf A[DC_];
    #pragma unroll
    for (int kk = 0; kk < DC_; ++kk) {
        const uint4* p = flatA + ((size_t)(rt * DC_ + kk) * 32 + lane) * 2;
        union { uint4 u[2]; v16bf v; } f;
        f.u[0] = p[0]; f.u[1] = p[1];
        A[kk] = f.v;
    }

    // --- GEMM: 8 col tiles x 8 K-chunks of v_wmma_f32_16x16x32_bf16 ---
    v8f acc[8];
    #pragma unroll
    for (int c = 0; c < 8; ++c) {
        const int ct = wave * 8 + c;
        v8f a = {0.f, 0.f, 0.f, 0.f, 0.f, 0.f, 0.f, 0.f};
        #pragma unroll
        for (int kk = 0; kk < DC_; ++kk) {
            const uint4* p = embB + ((size_t)(ct * DC_ + kk) * 32 + lane) * 2;
            union { uint4 u[2]; v16bf v; } f;
            f.u[0] = p[0]; f.u[1] = p[1];
            a = __builtin_amdgcn_wmma_f32_16x16x32_bf16(
                    false, A[kk], false, f.v, (short)0, a, false, false);
        }
        acc[c] = a;
    }
    __syncthreads();

    // --- logits (in-place in acc), kld partial, per-lane running max/argmax ---
    // C layout: VGPR r -> row m = r + 8*half, col = lcol (within tile)
    float kacc = 0.f;
    float bv[8]; int bc[8];
    #pragma unroll
    for (int r = 0; r < 8; ++r) { bv[r] = -3.4e38f; bc[r] = 0; }

    #pragma unroll
    for (int c = 0; c < 8; ++c) {
        const int ct = wave * 8 + c;
        const int k = ct * 16 + lcol;
        const float en = embNorm[k];
        #pragma unroll
        for (int r = 0; r < 8; ++r) {
            const int row = r + 8 * half;
            const int n = rowBase + row;
            const float dist = sRowN[row] + en - 2.f * acc[c][r];
            const float g = gumbel[(size_t)n * K_ + k];
            const float l = g - dist;                 // -d + gumbel
            const float p = 1.f / (1.f + __expf(dist)); // sigmoid(-d)
            kacc += p * __logf(fmaxf(p, 1e-8f));
            acc[c][r] = l;
            if (l > bv[r]) { bv[r] = l; bc[r] = k; }
        }
    }

    // --- cross-lane (16-wide halves) max/argmax, then cross-wave via LDS ---
    #pragma unroll
    for (int r = 0; r < 8; ++r) {
        float v = bv[r]; int cc = bc[r];
        #pragma unroll
        for (int off = 1; off < 16; off <<= 1) {
            float ov = __shfl_xor(v, off, 16);
            int   oc = __shfl_xor(cc, off, 16);
            if (ov > v) { v = ov; cc = oc; }
        }
        if (lcol == 0) {
            const int row = r + 8 * half;
            sMaxW[wave * 16 + row] = v;
            sAColW[wave * 16 + row] = cc;
        }
    }
    __syncthreads();
    if (tid < 16) {
        float m = -3.4e38f; int ci = 0;
        #pragma unroll
        for (int w = 0; w < 8; ++w) {
            float v = sMaxW[w * 16 + tid];
            if (v > m) { m = v; ci = sAColW[w * 16 + tid]; }
        }
        sMax[tid] = m; sIdx[tid] = ci;
    }
    __syncthreads();

    // --- exp-sum with global row max ---
    float sloc[8];
    #pragma unroll
    for (int r = 0; r < 8; ++r) sloc[r] = 0.f;
    #pragma unroll
    for (int c = 0; c < 8; ++c) {
        #pragma unroll
        for (int r = 0; r < 8; ++r) {
            const int row = r + 8 * half;
            sloc[r] += __expf(acc[c][r] - sMax[row]);
        }
    }
    #pragma unroll
    for (int r = 0; r < 8; ++r) {
        float s = sloc[r];
        #pragma unroll
        for (int off = 1; off < 16; off <<= 1) s += __shfl_xor(s, off, 16);
        if (lcol == 0) sSumW[wave * 16 + (r + 8 * half)] = s;
    }
    __syncthreads();
    if (tid < 16) {
        float s = 0.f;
        #pragma unroll
        for (int w = 0; w < 8; ++w) s += sSumW[w * 16 + tid];
        sSum[tid] = s;
    }
    __syncthreads();

    // --- encodings = (hard + y_soft) - y_soft ; record argmax info ---
    #pragma unroll
    for (int c = 0; c < 8; ++c) {
        const int ct = wave * 8 + c;
        const int k = ct * 16 + lcol;
        #pragma unroll
        for (int r = 0; r < 8; ++r) {
            const int row = r + 8 * half;
            const int n = rowBase + row;
            const float y = __expf(acc[c][r] - sMax[row]) / sSum[row];
            const float hard = (k == sIdx[row]) ? 1.f : 0.f;
            const float v = (hard + y) - y;
            enc[(size_t)n * K_ + k] = v;
            if (k == sIdx[row]) {
                idxArr[n] = k;
                vsel[n] = v;
                atomicAdd(&colsum[k], v);
            }
        }
    }

    // --- kld block reduction ---
    #pragma unroll
    for (int off = 16; off; off >>= 1) kacc += __shfl_xor(kacc, off, 32);
    if (lane == 0) sKld[wave] = kacc;
    __syncthreads();
    if (tid == 0) {
        float s = 0.f;
        #pragma unroll
        for (int w = 0; w < 8; ++w) s += sKld[w];
        atomicAdd(kldAcc, s);
    }
}

// ---------------- K4: quantized gather (NCHW) + e_latent_loss --------------
__global__ __launch_bounds__(256)
void k_quant(const float* __restrict__ emb, const float* __restrict__ x,
             const int* __restrict__ idxArr, const float* __restrict__ vsel,
             float* __restrict__ outQ, float* __restrict__ elatAcc) {
    const size_t o = (size_t)blockIdx.x * 256 + threadIdx.x;
    const int b   = (int)(o / ((size_t)C_ * HW_));
    const int rem = (int)(o % ((size_t)C_ * HW_));
    const int d   = rem / HW_;
    const int hw  = rem % HW_;
    const int n   = b * HW_ + hw;
    const float q = vsel[n] * emb[(size_t)idxArr[n] * D_ + d];
    outQ[o] = q;                       // coalesced NCHW store
    const float diff = q - x[o];
    float s = blockReduceSum256(diff * diff);
    if (threadIdx.x == 0) atomicAdd(elatAcc, s);
}

// ---------------- K5: loss + perplexity ------------------------------------
__global__ __launch_bounds__(1024)
void k_final(const float* __restrict__ colsum, const float* __restrict__ kldAcc,
             const float* __restrict__ elatAcc,
             float* __restrict__ outLoss, float* __restrict__ outPerp) {
    const int t = threadIdx.x;
    const float avg = colsum[t] * (1.f / (float)N_);
    float term = avg * __logf(avg + 1e-10f);
    #pragma unroll
    for (int off = 16; off; off >>= 1) term += __shfl_xor(term, off, 32);
    __shared__ float s[32];
    if ((t & 31) == 0) s[t >> 5] = term;
    __syncthreads();
    if (t == 0) {
        float tot = 0.f;
        #pragma unroll
        for (int i = 0; i < 32; ++i) tot += s[i];
        const float kld = *kldAcc / (float)N_;
        const float el  = *elatAcc / (float)((size_t)N_ * D_);
        *outLoss = 1.5f * (kld + el * (kld / fmaxf(el, 1e-8f)));
        *outPerp = __expf(-tot);
    }
}

// ---------------- host launcher --------------------------------------------
extern "C" void kernel_launch(void* const* d_in, const int* in_sizes, int n_in,
                              void* d_out, int out_size, void* d_ws, size_t ws_size,
                              hipStream_t stream) {
    const float* x   = (const float*)d_in[0];   // [B,C,H,W]
    const float* emb = (const float*)d_in[1];   // [K,D]
    const float* gum = (const float*)d_in[2];   // [N,K]
    float* out = (float*)d_out;

    // workspace carve (hipMalloc base is 256B aligned)
    unsigned short* flatA = (unsigned short*)d_ws;                    // 8388608 halves (16MB)
    unsigned short* embB  = flatA + (size_t)NT_ * DC_ * 32 * 16;      // 262144 halves (512KB)
    float* rowNorm = (float*)(embB + (size_t)CT_ * DC_ * 32 * 16);    // N_
    float* embNorm = rowNorm + N_;                                    // K_
    float* colsum  = embNorm + K_;                                    // K_
    float* kldAcc  = colsum + K_;                                     // 1
    float* elatAcc = kldAcc + 1;                                      // 1
    int*   idxArr  = (int*)(elatAcc + 1);                             // N_
    float* vsel    = (float*)(idxArr + N_);                           // N_

    // output layout: [loss][quantized NCHW][perplexity][encodings N,K]
    float* outLoss = out;
    float* outQ    = out + 1;
    float* outPerp = out + 1 + QELEMS_;
    float* outEnc  = out + 2 + QELEMS_;

    k_init<<<1, 1024, 0, stream>>>(colsum, kldAcc, elatAcc);
    k_prep_emb<<<K_, 256, 0, stream>>>(emb, embB, embNorm);
    k_prep_x<<<N_ / 256, 256, 0, stream>>>(x, flatA, rowNorm);
    k_dist_softmax<<<NT_, 256, 0, stream>>>((const uint4*)flatA, (const uint4*)embB,
                                            rowNorm, embNorm, gum,
                                            outEnc, colsum, kldAcc, idxArr, vsel);
    k_quant<<<(int)(QELEMS_ / 256), 256, 0, stream>>>(emb, x, idxArr, vsel, outQ, elatAcc);
    k_final<<<1, 1024, 0, stream>>>(colsum, kldAcc, elatAcc, outLoss, outPerp);
}
